// CausalSelfAttention_47004122087888
// MI455X (gfx1250) — matrix operand order repaired and checked by reading, not compiled, not run on hardware
//
#include <hip/hip_runtime.h>

// ---------------------------------------------------------------------------
// CDNA5 (gfx1250) causal self-attention: bf16 WMMA pipeline, fp32 accumulate.
// ---------------------------------------------------------------------------

typedef __attribute__((ext_vector_type(16))) __bf16        v16bf;
typedef __attribute__((ext_vector_type(8)))  float         v8f;
typedef __attribute__((ext_vector_type(8)))  unsigned int  v8u;

#define T_DIM 2048
#define C_DIM 2048
#define NHEAD 16
#define HDIM  128

// -------- helpers -----------------------------------------------------------

__device__ __forceinline__ unsigned short f2bf(float f) {
  unsigned int u = __builtin_bit_cast(unsigned int, f);
  u += 0x7fffu + ((u >> 16) & 1u);          // round-to-nearest-even
  return (unsigned short)(u >> 16);
}

// A-matrix fragment (16x32 bf16, row-major src with leading dim ld).
// Per ISA: lanes 0-15 -> M=lane, regs0-3 = K[kh*8 .. +7], regs4-7 = K[16+kh*8 .. +7]
__device__ __forceinline__ v16bf load_a_frag(const unsigned short* base, int ld,
                                             int row0, int k0) {
  int lane = threadIdx.x & 31;
  int m = lane & 15, kh = lane >> 4;
  const unsigned short* p = base + (size_t)(row0 + m) * ld + k0 + kh * 8;
  uint4 lo = *(const uint4*)p;          // K = k0+kh*8 .. +7   (16 bytes)
  uint4 hi = *(const uint4*)(p + 16);   // K = k0+16+kh*8 .. +7
  v8u u = { lo.x, lo.y, lo.z, lo.w, hi.x, hi.y, hi.z, hi.w };
  return __builtin_bit_cast(v16bf, u);
}

// B-matrix fragment (32x16 bf16) where B[k][n] = Wrow[n][k] (row-major weight).
// Per lane: n = lane&15, regs0-3 = K[kh*8 .. +7], regs4-7 = K[16+kh*8 .. +7].
__device__ __forceinline__ v16bf load_bt_frag(const unsigned short* base, int ld,
                                              int n0, int k0) {
  int lane = threadIdx.x & 31;
  int n = lane & 15, kh = lane >> 4;
  const unsigned short* p = base + (size_t)(n0 + n) * ld + k0 + kh * 8;
  uint4 lo = *(const uint4*)p;          // K = k0+kh*8 .. +7
  uint4 hi = *(const uint4*)(p + 16);   // K = k0+16+kh*8 .. +7
  v8u u = { lo.x, lo.y, lo.z, lo.w, hi.x, hi.y, hi.z, hi.w };
  return __builtin_bit_cast(v16bf, u);
}

__device__ __forceinline__ v8f wmma_bf16(v16bf a, v16bf b, v8f c) {
  return __builtin_amdgcn_wmma_f32_16x16x32_bf16(false, a, false, b,
                                                 (short)0, c, false, false);
}

// -------- kernel 1: fp32 -> bf16 conversion ---------------------------------

__global__ void cvt_bf16_kernel(const float4* __restrict__ src,
                                uint2* __restrict__ dst, int n4) {
  int i = blockIdx.x * blockDim.x + threadIdx.x;
  if (i >= n4) return;
  float4 f = src[i];
  uint2 o;
  o.x = (unsigned)f2bf(f.x) | ((unsigned)f2bf(f.y) << 16);
  o.y = (unsigned)f2bf(f.z) | ((unsigned)f2bf(f.w) << 16);
  dst[i] = o;
}

// -------- kernel 2: QKV projection GEMM -------------------------------------
// qkv[t][nn] = sum_k x[t][k] * W_attn[nn][k]; scatter into per-head q/k/v^T.
// Block = 256 threads (8 waves, 2x4), block tile 128x256, wave tile 64x64.

__global__ __launch_bounds__(256) void gemm_qkv_kernel(
    const unsigned short* __restrict__ xb,    // [T][C] bf16
    const unsigned short* __restrict__ wab,   // [3C][C] bf16
    unsigned short* __restrict__ qb,          // [H][T][D] bf16
    unsigned short* __restrict__ kb,          // [H][T][D] bf16
    unsigned short* __restrict__ vtb) {       // [H][D][T] bf16 (transposed!)
  int w = threadIdx.x >> 5;
  int lane = threadIdx.x & 31;
  int rowbase = blockIdx.x * 128 + (w >> 2) * 64;
  int colbase = blockIdx.y * 256 + (w & 3) * 64;

  v8f acc[4][4] = {};
  for (int k0 = 0; k0 < C_DIM; k0 += 32) {
    // prefetch next K tile into L2/L0 (global_prefetch_b8)
    if (k0 + 64 < C_DIM) {
      __builtin_prefetch(xb + (size_t)(rowbase + (lane & 15)) * C_DIM + k0 + 64, 0, 0);
      __builtin_prefetch(wab + (size_t)(colbase + (lane & 15)) * C_DIM + k0 + 64, 0, 0);
    }
    v16bf a[4], b[4];
#pragma unroll
    for (int i = 0; i < 4; i++) a[i] = load_a_frag(xb, C_DIM, rowbase + 16 * i, k0);
#pragma unroll
    for (int j = 0; j < 4; j++) b[j] = load_bt_frag(wab, C_DIM, colbase + 16 * j, k0);
#pragma unroll
    for (int i = 0; i < 4; i++)
#pragma unroll
      for (int j = 0; j < 4; j++) acc[i][j] = wmma_bf16(a[i], b[j], acc[i][j]);
  }

  int kh = lane >> 4, n = lane & 15;
#pragma unroll
  for (int i = 0; i < 4; i++)
#pragma unroll
    for (int j = 0; j < 4; j++)
#pragma unroll
      for (int r = 0; r < 8; r++) {
        int t  = rowbase + 16 * i + r + 8 * kh;          // D-frag row map
        int nn = colbase + 16 * j + n;                   // D-frag col map
        unsigned short val = f2bf(acc[i][j][r]);
        if (nn < C_DIM) {                                // Q
          int h = nn >> 7, d = nn & 127;
          qb[((size_t)(h * T_DIM) + t) * HDIM + d] = val;
        } else if (nn < 2 * C_DIM) {                     // K
          int c2 = nn - C_DIM, h = c2 >> 7, d = c2 & 127;
          kb[((size_t)(h * T_DIM) + t) * HDIM + d] = val;
        } else {                                         // V, stored transposed
          int c2 = nn - 2 * C_DIM, h = c2 >> 7, d = c2 & 127;
          vtb[((size_t)(h * HDIM) + d) * T_DIM + t] = val;
        }
      }
}

// -------- kernel 3: flash attention (one wave per head x 16-row q-block) ----

__global__ __launch_bounds__(128) void attn_kernel(
    const unsigned short* __restrict__ qb,    // [H][T][D]
    const unsigned short* __restrict__ kb,    // [H][T][D]
    const unsigned short* __restrict__ vtb,   // [H][D][T]
    unsigned short* __restrict__ ob) {        // att-out [T][C] bf16
  // P staged as bf16 already in A-fragment memory order: [row][k], row-major.
  __shared__ __align__(16) unsigned short plds[4][16][32];   // 4 KB
  int w = threadIdx.x >> 5;
  int lane = threadIdx.x & 31;
  int gw = blockIdx.x * 4 + w;                // 2048 waves total
  int h = gw >> 7;
  int qblk = gw & 127;
  int q0 = qblk * 16;

  const unsigned short* Q  = qb  + (size_t)h * T_DIM * HDIM;
  const unsigned short* Kp = kb  + (size_t)h * T_DIM * HDIM;
  const unsigned short* VT = vtb + (size_t)h * HDIM * T_DIM;

  v16bf qf[4];
#pragma unroll
  for (int c = 0; c < 4; c++) qf[c] = load_a_frag(Q, HDIM, q0, c * 32);

  v8f o[8] = {};
  float mrow[8], lrow[8];
#pragma unroll
  for (int r = 0; r < 8; r++) { mrow[r] = -__builtin_inff(); lrow[r] = 0.f; }

  int kh = lane >> 4, n = lane & 15;
  const float scale = 0.08838834764831845f;   // 1/sqrt(128)

  int nkb = (q0 >> 5) + 1;                    // 32-key blocks covering 0..q0+15
  for (int jb = 0; jb < nkb; jb++) {
    int kbase = jb * 32;
    // prefetch next key/value blocks (global_prefetch_b8)
    if (jb + 1 < nkb) {
      __builtin_prefetch(Kp + (size_t)(kbase + 32 + n) * HDIM, 0, 0);
      __builtin_prefetch(VT + (size_t)(n * 8) * T_DIM + kbase + 32, 0, 0);
    }
    // ---- S = Q @ K_blk^T (16 x 32, two 16x16 tiles) ----
    v8f s0 = {}, s1 = {};
#pragma unroll
    for (int c = 0; c < 4; c++) {
      v16bf b0 = load_bt_frag(Kp, HDIM, kbase,      c * 32);
      v16bf b1 = load_bt_frag(Kp, HDIM, kbase + 16, c * 32);
      s0 = wmma_bf16(qf[c], b0, s0);
      s1 = wmma_bf16(qf[c], b1, s1);
    }
    // ---- scale, causal mask, online softmax (row = r + 8*kh) ----
    float p0[8], p1[8];
#pragma unroll
    for (int r = 0; r < 8; r++) {
      int trow = q0 + r + 8 * kh;
      float v0 = (kbase + n      <= trow) ? s0[r] * scale : -__builtin_inff();
      float v1 = (kbase + 16 + n <= trow) ? s1[r] * scale : -__builtin_inff();
      float bm = fmaxf(v0, v1);
      bm = fmaxf(bm, __shfl_xor(bm, 1));
      bm = fmaxf(bm, __shfl_xor(bm, 2));
      bm = fmaxf(bm, __shfl_xor(bm, 4));
      bm = fmaxf(bm, __shfl_xor(bm, 8));
      float mnew = fmaxf(mrow[r], bm);
      float corr = __expf(mrow[r] - mnew);
      mrow[r] = mnew;
      float e0 = __expf(v0 - mnew);
      float e1 = __expf(v1 - mnew);
      float rs = e0 + e1;
      rs += __shfl_xor(rs, 1);
      rs += __shfl_xor(rs, 2);
      rs += __shfl_xor(rs, 4);
      rs += __shfl_xor(rs, 8);
      lrow[r] = lrow[r] * corr + rs;
#pragma unroll
      for (int c = 0; c < 8; c++) o[c][r] = o[c][r] * corr;
      p0[r] = e0; p1[r] = e1;
    }
    // ---- P: D-fragment layout -> A-fragment layout via LDS (bf16 staging) --
#pragma unroll
    for (int r = 0; r < 8; r++) {
      plds[w][r + 8 * kh][n]      = f2bf(p0[r]);
      plds[w][r + 8 * kh][16 + n] = f2bf(p1[r]);
    }
    asm volatile("s_wait_dscnt 0x0" ::: "memory");
    v16bf pf;
    {
      int row = lane & 15, kq = lane >> 4;
      uint4 lo = *(const uint4*)&plds[w][row][kq * 8];        // ds_load_b128
      uint4 hi = *(const uint4*)&plds[w][row][16 + kq * 8];   // ds_load_b128
      v8u u = { lo.x, lo.y, lo.z, lo.w, hi.x, hi.y, hi.z, hi.w };
      pf = __builtin_bit_cast(v16bf, u);
    }
    // ---- O += P @ V_blk : V^T gives contiguous loads in the key dimension --
#pragma unroll
    for (int c = 0; c < 8; c++) {
      v16bf vb = load_bt_frag(VT, T_DIM, c * 16, kbase);
      o[c] = wmma_bf16(pf, vb, o[c]);
    }
  }
  // ---- normalize and store att-out as bf16 [T][C], col = h*128 + d ----
#pragma unroll
  for (int r = 0; r < 8; r++) {
    int t = q0 + r + 8 * kh;
    float inv = 1.0f / lrow[r];
#pragma unroll
    for (int c = 0; c < 8; c++)
      ob[(size_t)t * C_DIM + h * HDIM + c * 16 + n] = f2bf(o[c][r] * inv);
  }
}

// -------- kernel 4: output projection GEMM (fp32 epilogue) ------------------

__global__ __launch_bounds__(256) void gemm_out_kernel(
    const unsigned short* __restrict__ ab,    // att-out [T][C] bf16
    const unsigned short* __restrict__ wpb,   // [C][C] bf16
    float* __restrict__ y) {                  // [T][C] fp32
  int w = threadIdx.x >> 5;
  int lane = threadIdx.x & 31;
  int rowbase = blockIdx.x * 128 + (w >> 2) * 64;
  int colbase = blockIdx.y * 256 + (w & 3) * 64;

  v8f acc[4][4] = {};
  for (int k0 = 0; k0 < C_DIM; k0 += 32) {
    if (k0 + 64 < C_DIM) {
      __builtin_prefetch(ab + (size_t)(rowbase + (lane & 15)) * C_DIM + k0 + 64, 0, 0);
      __builtin_prefetch(wpb + (size_t)(colbase + (lane & 15)) * C_DIM + k0 + 64, 0, 0);
    }
    v16bf a[4], b[4];
#pragma unroll
    for (int i = 0; i < 4; i++) a[i] = load_a_frag(ab, C_DIM, rowbase + 16 * i, k0);
#pragma unroll
    for (int j = 0; j < 4; j++) b[j] = load_bt_frag(wpb, C_DIM, colbase + 16 * j, k0);
#pragma unroll
    for (int i = 0; i < 4; i++)
#pragma unroll
      for (int j = 0; j < 4; j++) acc[i][j] = wmma_bf16(a[i], b[j], acc[i][j]);
  }

  int kh = lane >> 4, n = lane & 15;
#pragma unroll
  for (int i = 0; i < 4; i++)
#pragma unroll
    for (int j = 0; j < 4; j++)
#pragma unroll
      for (int r = 0; r < 8; r++) {
        int t  = rowbase + 16 * i + r + 8 * kh;
        int nn = colbase + 16 * j + n;
        y[(size_t)t * C_DIM + nn] = acc[i][j][r];
      }
}

// -------- launcher ----------------------------------------------------------

extern "C" void kernel_launch(void* const* d_in, const int* in_sizes, int n_in,
                              void* d_out, int out_size, void* d_ws, size_t ws_size,
                              hipStream_t stream) {
  (void)in_sizes; (void)n_in; (void)out_size; (void)ws_size;
  const float* x  = (const float*)d_in[0];   // [T][C]
  const float* Wa = (const float*)d_in[1];   // [3C][C]
  const float* Wp = (const float*)d_in[2];   // [C][C]
  float* y = (float*)d_out;                  // [T][C]

  unsigned short* ws  = (unsigned short*)d_ws;
  unsigned short* xb  = ws;                                       // T*C
  unsigned short* wab = xb  + (size_t)T_DIM * C_DIM;              // 3C*C
  unsigned short* wpb = wab + (size_t)3 * C_DIM * C_DIM;          // C*C
  unsigned short* qb  = wpb + (size_t)C_DIM * C_DIM;              // H*T*D
  unsigned short* kb  = qb  + (size_t)NHEAD * T_DIM * HDIM;
  unsigned short* vtb = kb  + (size_t)NHEAD * T_DIM * HDIM;
  unsigned short* ab  = vtb + (size_t)NHEAD * T_DIM * HDIM;       // T*C

  // fp32 -> bf16 conversions
  {
    int n4 = (T_DIM * C_DIM) / 4;
    cvt_bf16_kernel<<<(n4 + 255) / 256, 256, 0, stream>>>(
        (const float4*)x, (uint2*)xb, n4);
    int n4a = (3 * C_DIM * C_DIM) / 4;
    cvt_bf16_kernel<<<(n4a + 255) / 256, 256, 0, stream>>>(
        (const float4*)Wa, (uint2*)wab, n4a);
    int n4p = (C_DIM * C_DIM) / 4;
    cvt_bf16_kernel<<<(n4p + 255) / 256, 256, 0, stream>>>(
        (const float4*)Wp, (uint2*)wpb, n4p);
  }

  // QKV projection: M=2048, N=6144, K=2048
  gemm_qkv_kernel<<<dim3(T_DIM / 128, (3 * C_DIM) / 256), 256, 0, stream>>>(
      xb, wab, qb, kb, vtb);

  // Flash attention: 16 heads x 128 q-blocks = 2048 waves, 4 waves/block
  attn_kernel<<<(NHEAD * (T_DIM / 16)) / 4, 128, 0, stream>>>(qb, kb, vtb, ab);

  // Output projection: M=2048, N=2048, K=2048, fp32 out
  gemm_out_kernel<<<dim3(T_DIM / 128, C_DIM / 256), 256, 0, stream>>>(ab, wpb, y);
}
